// EPMoE_72138270703851
// MI455X (gfx1250) — matrix operand order repaired and checked by reading, not compiled, not run on hardware
//
#include <hip/hip_runtime.h>

#define T_TOK   256
#define N_EXP   32
#define TOPK    6
#define HID     1024
#define INT_DIM 768

typedef __attribute__((ext_vector_type(16))) __bf16 v16bf;
typedef __attribute__((ext_vector_type(2)))  __bf16 v2bf;
typedef __attribute__((ext_vector_type(8)))  float  v8f;

union Frag {
  v16bf v;
  unsigned int u[8];
  uint4 q[2];
};

// Hardware bf16 conversion (RNE). Prefer the packed cvt builtin when declared;
// otherwise plain fptrunc casts (lower to v_cvt_*bf16_f32 on gfx1250).
#if __has_builtin(__builtin_amdgcn_cvt_pk_bf16_f32)
__device__ __forceinline__ unsigned int pack2(float a, float b) {
  union { v2bf v; unsigned int u; } x;
  x.v = __builtin_amdgcn_cvt_pk_bf16_f32(a, b);
  return x.u;
}
#else
__device__ __forceinline__ unsigned int pack2(float a, float b) {
  union { __bf16 h[2]; unsigned int u; } x;
  x.h[0] = (__bf16)a; x.h[1] = (__bf16)b;
  return x.u;
}
#endif
__device__ __forceinline__ unsigned short f2bf(float f) {
  union { __bf16 h; unsigned short u; } x;
  x.h = (__bf16)f;
  return x.u;
}

// ---------------------------------------------------------------------------
// Kernel 1: biased grouped top-k routing (one thread per token).
// ---------------------------------------------------------------------------
__global__ __launch_bounds__(256) void moe_route(
    const float* __restrict__ logits, const float* __restrict__ bias,
    int* __restrict__ ids, float* __restrict__ wts)
{
  int t = blockIdx.x * blockDim.x + threadIdx.x;
  if (t >= T_TOK) return;
  float s[N_EXP], sc[N_EXP];
#pragma unroll
  for (int e = 0; e < N_EXP; ++e) {
    float l = logits[t * N_EXP + e];
    float sg = 1.0f / (1.0f + __expf(-l));
    s[e] = sg;
    sc[e] = sg + bias[e];
  }
  float gs[8];
#pragma unroll
  for (int g = 0; g < 8; ++g) {
    float m1 = -3.0e38f, m2 = -3.0e38f;
#pragma unroll
    for (int j = 0; j < 4; ++j) {
      float v = sc[g * 4 + j];
      if (v > m1) { m2 = m1; m1 = v; }
      else if (v > m2) { m2 = v; }
    }
    gs[g] = m1 + m2;
  }
  unsigned int gmask = 0;
#pragma unroll
  for (int k = 0; k < 4; ++k) {
    int bi = 0; float bv = -3.0e38f;
#pragma unroll
    for (int g = 0; g < 8; ++g)
      if (!((gmask >> g) & 1u) && gs[g] > bv) { bv = gs[g]; bi = g; }
    gmask |= (1u << bi);
  }
  unsigned int used = 0;
  int myid[TOPK]; float myw[TOPK]; float wsum = 0.0f;
#pragma unroll
  for (int k = 0; k < TOPK; ++k) {
    int bi = 0; float bv = -3.0e38f;
#pragma unroll
    for (int e = 0; e < N_EXP; ++e) {
      if (!((gmask >> (e >> 2)) & 1u)) continue;
      if ((used >> e) & 1u) continue;
      if (sc[e] > bv) { bv = sc[e]; bi = e; }
    }
    used |= (1u << bi);
    myid[k] = bi; myw[k] = s[bi]; wsum += s[bi];
  }
  float inv = 1.0f / wsum;
#pragma unroll
  for (int k = 0; k < TOPK; ++k) {
    ids[t * TOPK + k] = myid[k];
    wts[t * TOPK + k] = myw[k] * inv;
  }
}

// ---------------------------------------------------------------------------
// Kernel 2: deterministic per-expert token lists (sequential scan per expert).
// ---------------------------------------------------------------------------
__global__ void moe_build(const int* __restrict__ ids, const float* __restrict__ wts,
                          int* __restrict__ cnt, int* __restrict__ tok,
                          int* __restrict__ slot, float* __restrict__ wl)
{
  int e = blockIdx.x;
  if (threadIdx.x != 0) return;
  int c = 0;
  for (int t = 0; t < T_TOK; ++t)
    for (int k = 0; k < TOPK; ++k)
      if (ids[t * TOPK + k] == e) {
        tok[e * T_TOK + c] = t;
        slot[e * T_TOK + c] = k;
        wl[e * T_TOK + c] = wts[t * TOPK + k];
        ++c;
      }
  cnt[e] = c;
}

// ---------------------------------------------------------------------------
// Kernel 3: hidden_states fp32 -> bf16 once (A-operand; L2-resident after).
// ---------------------------------------------------------------------------
__global__ __launch_bounds__(256) void moe_cvt_hs(
    const float* __restrict__ hs, unsigned short* __restrict__ hsb)
{
  int idx = blockIdx.x * 256 + threadIdx.x;  // float4 index, 65536 total
  float4 f = ((const float4*)hs)[idx];
  uint2 v; v.x = pack2(f.x, f.y); v.y = pack2(f.z, f.w);
  ((uint2*)hsb)[idx] = v;
}

// ---------------------------------------------------------------------------
// Kernel 4: grouped GEMM1 gu = hs @ w13^T, SiLU(gate)*up -> act (bf16).
// grid (expert, 4 M-tiles of 64 tokens, 6 N-tiles of 128 gate cols), 8 waves.
// All 4 A-fragments are loaded before the 8-WMMA burst so the scheduler can
// wait once and issue WMMAs back-to-back.
// ---------------------------------------------------------------------------
__global__ __launch_bounds__(256) void moe_gemm1(
    const unsigned short* __restrict__ hsb, const float* __restrict__ w13,
    const int* __restrict__ cnt, const int* __restrict__ tok,
    unsigned short* __restrict__ act)
{
  const int e = blockIdx.x, tT = blockIdx.y, nT = blockIdx.z;
  const int c = cnt[e];
  if (tT * 64 >= c) return;
  const int wv = threadIdx.x >> 5;
  const int lane = threadIdx.x & 31;
  const int half = lane >> 4;
  const int m = lane & 15;

  int tk[4];
#pragma unroll
  for (int s = 0; s < 4; ++s) {
    int r = tT * 64 + s * 16 + m;
    tk[s] = (r < c) ? tok[e * T_TOK + r] : tok[e * T_TOK];  // clamp; masked on store
  }
  const int col = nT * 128 + wv * 16 + m;                   // gate column (N)
  const float* bg = w13 + ((size_t)e * (2 * INT_DIM) + col) * HID;
  const float* bu = bg + (size_t)INT_DIM * HID;

  v8f accG[4], accU[4];
#pragma unroll
  for (int s = 0; s < 4; ++s) { accG[s] = (v8f){0,0,0,0,0,0,0,0}; accU[s] = (v8f){0,0,0,0,0,0,0,0}; }

  for (int k0 = 0; k0 < HID; k0 += 32) {
    // A fragments first (L2-resident bf16), so loads overlap the cvt VALU work
    Frag a[4];
#pragma unroll
    for (int s = 0; s < 4; ++s) {
      const unsigned short* pa = hsb + (size_t)tk[s] * HID + k0 + half * 16;
      a[s].q[0] = *(const uint4*)pa;
      a[s].q[1] = *(const uint4*)(pa + 8);
    }
    // B fragments: stream fp32 weights, pack to bf16 in-register
    Frag fg, fu;
    const float4* pg = (const float4*)(bg + k0 + half * 16);
    const float4* pu = (const float4*)(bu + k0 + half * 16);
#pragma unroll
    for (int g2 = 0; g2 < 4; ++g2) {
      float4 f = pg[g2];
      fg.u[2 * g2]     = pack2(f.x, f.y);
      fg.u[2 * g2 + 1] = pack2(f.z, f.w);
      float4 q = pu[g2];
      fu.u[2 * g2]     = pack2(q.x, q.y);
      fu.u[2 * g2 + 1] = pack2(q.z, q.w);
    }
    if (k0 + 96 < HID) {  // pull upcoming weight lines toward the WGP
      __builtin_prefetch(bg + k0 + 96 + half * 16, 0, 3);
      __builtin_prefetch(bu + k0 + 96 + half * 16, 0, 3);
    }
#pragma unroll
    for (int s = 0; s < 4; ++s) {
      accG[s] = __builtin_amdgcn_wmma_f32_16x16x32_bf16(false, a[s].v, false, fg.v,
                                                        (short)0, accG[s], false, false);
      accU[s] = __builtin_amdgcn_wmma_f32_16x16x32_bf16(false, a[s].v, false, fu.v,
                                                        (short)0, accU[s], false, false);
    }
  }
#pragma unroll
  for (int s = 0; s < 4; ++s) {
#pragma unroll
    for (int r = 0; r < 8; ++r) {
      int trow = tT * 64 + s * 16 + r + half * 8;
      if (trow < c) {
        float g = accG[s][r], u = accU[s][r];
        float av = g * (1.0f / (1.0f + __expf(-g))) * u;   // silu(g)*u
        act[((size_t)e * T_TOK + trow) * INT_DIM + col] = f2bf(av);
      }
    }
  }
}

// ---------------------------------------------------------------------------
// Kernel 5: grouped GEMM2 y = act @ w2^T, scaled by combine weight, scattered
// into per-(token, topk-slot) buffer. grid (expert, 4 M-tiles, 8 N-tiles).
// ---------------------------------------------------------------------------
__global__ __launch_bounds__(256) void moe_gemm2(
    const unsigned short* __restrict__ act, const float* __restrict__ w2,
    const int* __restrict__ cnt, const int* __restrict__ tok,
    const int* __restrict__ slot, const float* __restrict__ wl,
    float* __restrict__ slots)
{
  const int e = blockIdx.x, tT = blockIdx.y, nT = blockIdx.z;
  const int c = cnt[e];
  if (tT * 64 >= c) return;
  const int wv = threadIdx.x >> 5;
  const int lane = threadIdx.x & 31;
  const int half = lane >> 4;
  const int m = lane & 15;

  const int col = nT * 128 + wv * 16 + m;                   // output H column (N)
  const float* bw = w2 + ((size_t)e * HID + col) * INT_DIM;
  const unsigned short* abase = act + ((size_t)e * T_TOK + tT * 64) * INT_DIM;

  v8f acc[4];
#pragma unroll
  for (int s = 0; s < 4; ++s) acc[s] = (v8f){0,0,0,0,0,0,0,0};

  for (int k0 = 0; k0 < INT_DIM; k0 += 32) {
    Frag a[4];
#pragma unroll
    for (int s = 0; s < 4; ++s) {
      const unsigned short* pa = abase + (size_t)(s * 16 + m) * INT_DIM + k0 + half * 16;
      a[s].q[0] = *(const uint4*)pa;
      a[s].q[1] = *(const uint4*)(pa + 8);
    }
    Frag fb;
    const float4* pb = (const float4*)(bw + k0 + half * 16);
#pragma unroll
    for (int g2 = 0; g2 < 4; ++g2) {
      float4 f = pb[g2];
      fb.u[2 * g2]     = pack2(f.x, f.y);
      fb.u[2 * g2 + 1] = pack2(f.z, f.w);
    }
    if (k0 + 96 < INT_DIM)
      __builtin_prefetch(bw + k0 + 96 + half * 16, 0, 3);
#pragma unroll
    for (int s = 0; s < 4; ++s) {
      acc[s] = __builtin_amdgcn_wmma_f32_16x16x32_bf16(false, a[s].v, false, fb.v,
                                                       (short)0, acc[s], false, false);
    }
  }
#pragma unroll
  for (int s = 0; s < 4; ++s) {
#pragma unroll
    for (int r = 0; r < 8; ++r) {
      int trow = tT * 64 + s * 16 + r + half * 8;
      if (trow < c) {
        int   tkn = tok[e * T_TOK + trow];
        int   sl  = slot[e * T_TOK + trow];
        float w   = wl[e * T_TOK + trow];
        slots[((size_t)tkn * TOPK + sl) * HID + col] = w * acc[s][r];
      }
    }
  }
}

// ---------------------------------------------------------------------------
// Kernel 6: fixed-order combine over the 6 slots (deterministic, no atomics).
// ---------------------------------------------------------------------------
__global__ __launch_bounds__(256) void moe_combine(
    const float* __restrict__ slots, float* __restrict__ out)
{
  int idx = blockIdx.x * 256 + threadIdx.x;   // < 256*1024
  int t = idx >> 10, h = idx & 1023;
  float s = 0.0f;
#pragma unroll
  for (int k = 0; k < TOPK; ++k)
    s += slots[((size_t)t * TOPK + k) * HID + h];
  out[idx] = s;
}

// ---------------------------------------------------------------------------
// Workspace layout (bytes)
// ---------------------------------------------------------------------------
static constexpr size_t OFF_IDS   = 0;                         // int  [256*6]
static constexpr size_t OFF_WTS   = 6144;                      // f32  [256*6]
static constexpr size_t OFF_CNT   = 12288;                     // int  [32]
static constexpr size_t OFF_TOK   = 12544;                     // int  [32*256]
static constexpr size_t OFF_SLOT  = 45312;                     // int  [32*256]
static constexpr size_t OFF_WL    = 78080;                     // f32  [32*256]
static constexpr size_t OFF_HSB   = 110848;                    // bf16 [256*1024]
static constexpr size_t OFF_ACT   = 635904;                    // bf16 [32*256*768]
static constexpr size_t OFF_SLOTS = 13218816;                  // f32  [256*6*1024]

extern "C" void kernel_launch(void* const* d_in, const int* in_sizes, int n_in,
                              void* d_out, int out_size, void* d_ws, size_t ws_size,
                              hipStream_t stream)
{
  const float* hs  = (const float*)d_in[0];   // [256,1024]
  const float* rl  = (const float*)d_in[1];   // [256,32]
  const float* cb  = (const float*)d_in[2];   // [32]
  const float* w13 = (const float*)d_in[3];   // [32,1536,1024]
  const float* w2  = (const float*)d_in[4];   // [32,1024,768]
  float* out = (float*)d_out;

  char* ws = (char*)d_ws;
  int*            ids   = (int*)(ws + OFF_IDS);
  float*          wts   = (float*)(ws + OFF_WTS);
  int*            cnt   = (int*)(ws + OFF_CNT);
  int*            tok   = (int*)(ws + OFF_TOK);
  int*            slot  = (int*)(ws + OFF_SLOT);
  float*          wl    = (float*)(ws + OFF_WL);
  unsigned short* hsb   = (unsigned short*)(ws + OFF_HSB);
  unsigned short* act   = (unsigned short*)(ws + OFF_ACT);
  float*          slots = (float*)(ws + OFF_SLOTS);

  moe_route  <<<1,   256, 0, stream>>>(rl, cb, ids, wts);
  moe_build  <<<32,  32,  0, stream>>>(ids, wts, cnt, tok, slot, wl);
  moe_cvt_hs <<<256, 256, 0, stream>>>(hs, hsb);
  moe_gemm1  <<<dim3(32, 4, 6), 256, 0, stream>>>(hsb, w13, cnt, tok, act);
  moe_gemm2  <<<dim3(32, 4, 8), 256, 0, stream>>>(act, w2, cnt, tok, slot, wl, slots);
  moe_combine<<<1024, 256, 0, stream>>>(slots, out);
}
